// HybridForecaster_89060441850413
// MI455X (gfx1250) — compile-verified, matrix operand order checked by reading, not run on hardware
//
#include <hip/hip_runtime.h>
#include <cstdint>
#include <cstddef>

// ---------------------------------------------------------------------------
// HybridForecaster for MI455X (gfx1250, wave32, WMMA bf16 + TDM path)
// B=64 T=2048 C=64 H=256 HOR=96
// ---------------------------------------------------------------------------

#define B_   64
#define T_   2048
#define C_   64
#define H_   256
#define HOR_ 96
#define CH_  320   // C + H
#define G3_  768   // 3*H

typedef __bf16 bf16;
typedef bf16  bf16x8  __attribute__((ext_vector_type(8)));
typedef bf16  bf16x16 __attribute__((ext_vector_type(16)));
typedef float f32x8   __attribute__((ext_vector_type(8)));
typedef unsigned int u32x4 __attribute__((ext_vector_type(4)));
typedef int   i32x4 __attribute__((ext_vector_type(4)));
typedef int   i32x8 __attribute__((ext_vector_type(8)));

__device__ __forceinline__ bf16 f2bf(float f) {
    uint32_t u = __builtin_bit_cast(uint32_t, f);
    u += 0x7FFFu + ((u >> 16) & 1u);           // round-to-nearest-even
    unsigned short h = (unsigned short)(u >> 16);
    return __builtin_bit_cast(bf16, h);
}
__device__ __forceinline__ float bf2f(bf16 x) {
    uint32_t u = ((uint32_t)__builtin_bit_cast(unsigned short, x)) << 16;
    return __builtin_bit_cast(float, u);
}
__device__ __forceinline__ float sigm(float x) { return 1.0f / (1.0f + __expf(-x)); }

// Load one 16x32 bf16 WMMA operand tile from a row-major [rows][ld] matrix.
// lane(0-15)=row, half=lane>>4 selects K=8*half.. ; two b128 loads per lane.
__device__ __forceinline__ bf16x16 load_tile(const bf16* p, int ld, int lane) {
    int row  = lane & 15;
    int half = lane >> 4;
    const bf16* q = p + (size_t)row * ld + (half << 3);
    bf16x8 lo = *reinterpret_cast<const bf16x8*>(q);
    bf16x8 hi = *reinterpret_cast<const bf16x8*>(q + 16);
    return __builtin_shufflevector(lo, hi, 0,1,2,3,4,5,6,7,8,9,10,11,12,13,14,15);
}

__device__ __forceinline__ f32x8 wmma_bf16(bf16x16 a, bf16x16 b, f32x8 c) {
    return __builtin_amdgcn_wmma_f32_16x16x32_bf16(false, a, false, b, (short)0, c,
                                                   false, false);
}
__device__ __forceinline__ f32x8 zero8() {
    f32x8 z = {0.f,0.f,0.f,0.f,0.f,0.f,0.f,0.f};
    return z;
}

// ---------------------------------------------------------------------------
// Tensor Data Mover: 1-D contiguous DMA, global -> LDS[lds_off], nbytes%8==0.
// D# per CDNA5 ISA (group0 128b, group1 256b).
// This toolchain exposes the 6-arg builtin: (v4u, v8i, v4i, v4i, v8i, cpol).
// ---------------------------------------------------------------------------
__device__ __forceinline__ void tdm_load_1d(uint32_t lds_off, const void* gaddr,
                                            uint32_t nbytes) {
    uint64_t ga    = (uint64_t)(uintptr_t)gaddr;
    uint32_t elems = nbytes >> 3;                       // 8-byte elements
    u32x4 g0;
    g0[0] = 1u;                                         // count=1 (valid), user mode
    g0[1] = lds_off;                                    // lds_addr (bytes)
    g0[2] = (uint32_t)ga;                               // global_addr[31:0]
    g0[3] = (uint32_t)((ga >> 32) & 0x01FFFFFFu)        // global_addr[56:32]
          | (2u << 30);                                 // type=2 ("image")
    i32x8 g1;
    g1[0] = (int)(3u << 16);                            // data_size=3 (8B), mask=0
    g1[1] = (int)((elems & 0xFFFFu) << 16);             // tensor_dim0[15:0]
    g1[2] = (int)(((elems >> 16) & 0xFFFFu)             // tensor_dim0[31:16]
          | (1u << 16));                                // tensor_dim1 = 1
    g1[3] = (int)((elems & 0xFFFFu) << 16);             // tile_dim0 = elems
    g1[4] = 0;                                          // tile_dim1/2 unused
    g1[5] = (int)elems;                                 // tensor_dim0_stride[31:0]
    g1[6] = 0;
    g1[7] = 0;
    i32x4 gz4  = {0, 0, 0, 0};
    i32x8 gz8  = {0, 0, 0, 0, 0, 0, 0, 0};
    __builtin_amdgcn_tensor_load_to_lds(g0, g1, gz4, gz4, gz8, 0);
}

// ---------------------------------------------------------------------------
// 1) fp32 -> bf16 weight conversion
// ---------------------------------------------------------------------------
__global__ void cvt_bf16_k(const float* __restrict__ src, bf16* __restrict__ dst, int n) {
    int i = blockIdx.x * 256 + threadIdx.x;
    if (i < n) dst[i] = f2bf(src[i]);
}

// x [B*T, 64] -> first 64 columns of hcomb [B*T, 320] (bf16)
__global__ void pack_x_k(const float* __restrict__ x, bf16* __restrict__ hcomb) {
    int i = blockIdx.x * 256 + threadIdx.x;     // < B*T*C
    int m = i >> 6, c = i & 63;
    hcomb[(size_t)m * CH_ + c] = f2bf(x[i]);
}

// ---------------------------------------------------------------------------
// 2) SNN input current: cur[B*T,256] = x @ snn_w^T + snn_b   (K=64)
// ---------------------------------------------------------------------------
__global__ void snn_cur_gemm(const bf16* __restrict__ hcomb, const bf16* __restrict__ wsn,
                             const float* __restrict__ snn_b, bf16* __restrict__ cur) {
    int lane = threadIdx.x & 31;
    int wid  = blockIdx.x * 8 + (threadIdx.x >> 5);
    int mt    = wid >> 2;           // 0..8191
    int nbase = (wid & 3) << 6;     // 0,64,128,192
    f32x8 acc[4];
    #pragma unroll
    for (int i = 0; i < 4; ++i) acc[i] = zero8();
    const bf16* abase = hcomb + (size_t)(mt * 16) * CH_;
    #pragma unroll
    for (int k0 = 0; k0 < C_; k0 += 32) {
        bf16x16 a = load_tile(abase + k0, CH_, lane);
        #pragma unroll
        for (int i = 0; i < 4; ++i) {
            bf16x16 b = load_tile(wsn + (size_t)(nbase + i * 16) * C_ + k0, C_, lane);
            acc[i] = wmma_bf16(a, b, acc[i]);
        }
    }
    int n = lane & 15, half = lane >> 4;
    #pragma unroll
    for (int i = 0; i < 4; ++i) {
        int col = nbase + i * 16 + n;
        float bias = snn_b[col];
        #pragma unroll
        for (int r = 0; r < 8; ++r) {
            int row = mt * 16 + r + 8 * half;
            cur[(size_t)row * H_ + col] = f2bf(acc[i][r] + bias);
        }
    }
}

// ---------------------------------------------------------------------------
// 3) LIF scan: sequential over T, parallel over B*H = 16384 threads
// ---------------------------------------------------------------------------
__global__ void lif_scan(const bf16* __restrict__ cur, bf16* __restrict__ hcomb) {
    int idx = blockIdx.x * blockDim.x + threadIdx.x;   // 0..16383
    int b = idx >> 8, h = idx & 255;
    const bf16* cp = cur   + (size_t)b * T_ * H_  + h;
    bf16*       sp = hcomb + (size_t)b * T_ * CH_ + C_ + h;
    float mem = 0.0f;
    #pragma unroll 4
    for (int t = 0; t < T_; ++t) {
        float i_t   = bf2f(cp[(size_t)t * H_]);
        float reset = (mem > 1.0f) ? 1.0f : 0.0f;      // detached reset (prev mem)
        mem = 0.9f * mem + i_t - reset;
        sp[(size_t)t * CH_] = f2bf((mem > 1.0f) ? 1.0f : 0.0f);
    }
}

// ---------------------------------------------------------------------------
// 4) GRU input gates: gx = hcomb @ wih^T + bih, K=320, N=768
//    output layout [T, B, 3H] (bf16): per-t slab contiguous for the TDM
// ---------------------------------------------------------------------------
__global__ void gx_gemm(const bf16* __restrict__ hcomb, const bf16* __restrict__ wih,
                        const float* __restrict__ bih, bf16* __restrict__ gx) {
    int lane = threadIdx.x & 31;
    int wid  = blockIdx.x * 8 + (threadIdx.x >> 5);
    int mt    = wid / 12;           // 0..8191
    int nbase = (wid % 12) << 6;    // 0..704
    f32x8 acc[4];
    #pragma unroll
    for (int i = 0; i < 4; ++i) acc[i] = zero8();
    const bf16* abase = hcomb + (size_t)(mt * 16) * CH_;
    #pragma unroll 2
    for (int k0 = 0; k0 < CH_; k0 += 32) {
        __builtin_prefetch(abase + k0 + 64, 0, 0);     // global_prefetch_b8
        bf16x16 a = load_tile(abase + k0, CH_, lane);
        #pragma unroll
        for (int i = 0; i < 4; ++i) {
            bf16x16 b = load_tile(wih + (size_t)(nbase + i * 16) * CH_ + k0, CH_, lane);
            acc[i] = wmma_bf16(a, b, acc[i]);
        }
    }
    int n = lane & 15, half = lane >> 4;
    #pragma unroll
    for (int i = 0; i < 4; ++i) {
        int col = nbase + i * 16 + n;
        float bias = bih[col];
        #pragma unroll
        for (int r = 0; r < 8; ++r) {
            int row = mt * 16 + r + 8 * half;      // row = b*T + t
            int bb = row >> 11, tt = row & (T_ - 1);
            gx[((size_t)tt * B_ + bb) * G3_ + col] = f2bf(acc[i][r] + bias);
        }
    }
}

// ---------------------------------------------------------------------------
// 5) Persistent GRU scan + fused head GEMM. One workgroup, 32 waves.
//    - h (f32) lives in REGISTERS (each element has a unique owner thread);
//      bf16 mirror in LDS is the WMMA A operand.          LDS: 32 KB
//    - gx_t slab (96 KB) DMA'd into LDS by the Tensor Data Mover, issued by
//      wave 0 one step ahead, overlapped with the WMMA phase. LDS: 96 KB
// ---------------------------------------------------------------------------
__global__ __launch_bounds__(1024) void gru_scan_head(
    const bf16* __restrict__ gx, const bf16* __restrict__ whh,
    const float* __restrict__ bhh, const bf16* __restrict__ headw,
    const float* __restrict__ headb, float* __restrict__ out) {

    __shared__ __align__(16) bf16 hB[B_ * H_];       // 32 KB
    __shared__ __align__(16) bf16 gxs[B_ * G3_];     // 96 KB staging for gx_t

    int tid = threadIdx.x, lane = tid & 31, wave = tid >> 5;
    for (int i = tid; i < B_ * H_; i += 1024) hB[i] = f2bf(0.0f);

    uint32_t gxs_off = (uint32_t)(uintptr_t)(&gxs[0]);   // LDS byte offset
    if (wave == 0) tdm_load_1d(gxs_off, gx, B_ * G3_ * 2);   // slab t=0

    float hreg[2][8];
    #pragma unroll
    for (int g = 0; g < 2; ++g)
        #pragma unroll
        for (int r = 0; r < 8; ++r) hreg[g][r] = 0.0f;

    int n = lane & 15, half = lane >> 4;
    int jt  = wave >> 1;               // 0..15
    int mt0 = (wave & 1) << 1;         // 0 or 2
    int j = jt * 16 + n;
    float bR = bhh[j], bZ = bhh[H_ + j], bN = bhh[2 * H_ + j];
    const bf16* wR = whh + (size_t)(jt * 16) * H_;
    const bf16* wZ = whh + (size_t)((jt + 16) * 16) * H_;
    const bf16* wN = whh + (size_t)((jt + 32) * 16) * H_;

    __syncthreads();   // hB initialized

    for (int t = 0; t < T_; ++t) {
        f32x8 accR[2], accZ[2], accN[2];
        #pragma unroll
        for (int g = 0; g < 2; ++g) { accR[g] = zero8(); accZ[g] = zero8(); accN[g] = zero8(); }

        #pragma unroll
        for (int ks = 0; ks < 8; ++ks) {
            int k0 = ks * 32;
            bf16x16 a0  = load_tile(hB + (size_t)(mt0 * 16) * H_ + k0, H_, lane);  // ds_load_b128
            bf16x16 a1  = load_tile(hB + (size_t)((mt0 + 1) * 16) * H_ + k0, H_, lane);
            bf16x16 bRt = load_tile(wR + k0, H_, lane);
            bf16x16 bZt = load_tile(wZ + k0, H_, lane);
            bf16x16 bNt = load_tile(wN + k0, H_, lane);
            accR[0] = wmma_bf16(a0, bRt, accR[0]);  accR[1] = wmma_bf16(a1, bRt, accR[1]);
            accZ[0] = wmma_bf16(a0, bZt, accZ[0]);  accZ[1] = wmma_bf16(a1, bZt, accZ[1]);
            accN[0] = wmma_bf16(a0, bNt, accN[0]);  accN[1] = wmma_bf16(a1, bNt, accN[1]);
        }

        if (wave == 0) __builtin_amdgcn_s_wait_tensorcnt(0);   // gx_t slab landed
        __syncthreads();   // hB reads done + gxs published to all waves

        #pragma unroll
        for (int g = 0; g < 2; ++g) {
            int mt = mt0 + g;
            #pragma unroll
            for (int r = 0; r < 8; ++r) {
                int m = mt * 16 + r + 8 * half;
                const bf16* gp = gxs + (size_t)m * G3_ + j;    // ds_load_u16
                float xr = bf2f(gp[0]);
                float xz = bf2f(gp[H_]);
                float xn = bf2f(gp[2 * H_]);
                float rg = sigm(xr + accR[g][r] + bR);
                float zg = sigm(xz + accZ[g][r] + bZ);
                float ng = tanhf(xn + rg * (accN[g][r] + bN));
                float hnew = (1.0f - zg) * ng + zg * hreg[g][r];
                hreg[g][r] = hnew;
                hB[m * H_ + j] = f2bf(hnew);
            }
        }
        __syncthreads();   // h(t) published; gxs free for next slab

        if (wave == 0 && t + 1 < T_)
            tdm_load_1d(gxs_off, gx + (size_t)(t + 1) * B_ * G3_, B_ * G3_ * 2);
    }

    // ---- fused head: out[64,96] = hT @ head_w^T + head_b (24 tiles, K=256) ----
    if (wave < 24) {
        int mt = wave / 6, nt = wave % 6;
        f32x8 acc = zero8();
        #pragma unroll
        for (int ks = 0; ks < 8; ++ks) {
            int k0 = ks * 32;
            bf16x16 a = load_tile(hB + (size_t)(mt * 16) * H_ + k0, H_, lane);
            bf16x16 b = load_tile(headw + (size_t)(nt * 16) * H_ + k0, H_, lane);
            acc = wmma_bf16(a, b, acc);
        }
        int col = nt * 16 + n;
        float bias = headb[col];
        #pragma unroll
        for (int r = 0; r < 8; ++r) {
            int row = mt * 16 + r + 8 * half;
            out[row * HOR_ + col] = acc[r] + bias;
        }
    }
}

// ---------------------------------------------------------------------------
// launch
// ---------------------------------------------------------------------------
extern "C" void kernel_launch(void* const* d_in, const int* in_sizes, int n_in,
                              void* d_out, int out_size, void* d_ws, size_t ws_size,
                              hipStream_t stream) {
    const float* x      = (const float*)d_in[0];
    const float* snn_w  = (const float*)d_in[1];
    const float* snn_b  = (const float*)d_in[2];
    const float* wih    = (const float*)d_in[3];
    const float* whh    = (const float*)d_in[4];
    const float* bih    = (const float*)d_in[5];
    const float* bhh    = (const float*)d_in[6];
    const float* head_w = (const float*)d_in[7];
    const float* head_b = (const float*)d_in[8];
    float* out = (float*)d_out;

    char* ws = (char*)d_ws;
    size_t off = 0;
    bf16* hcomb   = (bf16*)(ws + off); off += (size_t)B_ * T_ * CH_ * 2;   // 84 MB
    bf16* cur     = (bf16*)(ws + off); off += (size_t)B_ * T_ * H_  * 2;   // 67 MB
    bf16* gx      = (bf16*)(ws + off); off += (size_t)B_ * T_ * G3_ * 2;   // 201 MB
    bf16* snn_wh  = (bf16*)(ws + off); off += (size_t)H_ * C_ * 2;
    bf16* wih_h   = (bf16*)(ws + off); off += (size_t)G3_ * CH_ * 2;
    bf16* whh_h   = (bf16*)(ws + off); off += (size_t)G3_ * H_ * 2;
    bf16* head_wh = (bf16*)(ws + off); off += (size_t)HOR_ * H_ * 2;
    (void)off; (void)ws_size; (void)in_sizes; (void)n_in; (void)out_size;

    // weight conversions
    cvt_bf16_k<<<(H_ * C_ + 255) / 256, 256, 0, stream>>>(snn_w, snn_wh, H_ * C_);
    cvt_bf16_k<<<(G3_ * CH_ + 255) / 256, 256, 0, stream>>>(wih, wih_h, G3_ * CH_);
    cvt_bf16_k<<<(G3_ * H_ + 255) / 256, 256, 0, stream>>>(whh, whh_h, G3_ * H_);
    cvt_bf16_k<<<(HOR_ * H_ + 255) / 256, 256, 0, stream>>>(head_w, head_wh, HOR_ * H_);

    // x -> hcomb[:, 0:64]
    pack_x_k<<<(B_ * T_ * C_) / 256, 256, 0, stream>>>(x, hcomb);

    // SNN currents
    snn_cur_gemm<<<4096, 256, 0, stream>>>(hcomb, snn_wh, snn_b, cur);

    // LIF membrane scan -> spikes into hcomb[:, 64:320]
    lif_scan<<<64, 256, 0, stream>>>(cur, hcomb);

    // GRU input-gate GEMM -> gx [T,B,3H]
    gx_gemm<<<12288, 256, 0, stream>>>(hcomb, wih_h, bih, gx);

    // persistent GRU scan (TDM-fed) + head
    gru_scan_head<<<1, 1024, 0, stream>>>(gx, whh_h, bhh, head_wh, head_b, out);
}
